// Model_30485677867035
// MI455X (gfx1250) — matrix operand order repaired and checked by reading, not compiled, not run on hardware
//
#include <hip/hip_runtime.h>
#include <hip/hip_bf16.h>

// ---------------- problem constants (match reference) ----------------
#define B_   64
#define L_   2048
#define C_   128
#define H_   128
#define K_   64
#define P_   336
#define NBLK 4
#define EPSF 1e-5f

typedef __attribute__((ext_vector_type(16))) _Float16 v16h;
typedef __attribute__((ext_vector_type(8)))  _Float16 v8h;
typedef __attribute__((ext_vector_type(4)))  _Float16 v4h;
typedef __attribute__((ext_vector_type(8)))  float    v8f;
typedef long long ll;

#define ACT_NONE 0
#define ACT_RELU 1
#define ACT_GELU 2

// =====================================================================
// Tiled WMMA GEMM:  D[b] = epilogue( alpha * A[b] @ B[b] + bias )
//   A element (m,k) at A[bA*b + rA*m + cA*k]   (lets us feed swapaxes views)
//   B element (k,n) at Bm[bB*b + kB*k + nB*n]  (kB=1,nB=ld -> implicit B^T)
//   D element (m,n) at D[bD*b + rD*m + cD*n]
// epilogue: v = alpha*acc + bias[n]; v = act(v); v += Res[off];
//           v = v*scale[b*M+m] + shift[b*M+m]
// Block = 128 threads = 4 waves; tile 64x64; k-step 32.
// LDS layouts chosen so every WMMA fragment is 2x ds_load_b128:
//   As [64][48]  row-major, lane fragment = K[8h,8h+8) and K[16+8h,+8)
//   BsT[64][48]  B stored transposed [n][k], fragment = K[8h,8h+8) and K[16+8h,+8)
// NOTE: requires Kd % 4 == 0 (all call sites use Kd in {128, 2048}).
// =====================================================================
template<int ACT>
__global__ __launch_bounds__(128) void gemm_wmma(
    const float* __restrict__ A, const float* __restrict__ Bm,
    const float* __restrict__ bias, const float* __restrict__ Res,
    const float* __restrict__ scale, const float* __restrict__ shift,
    float* __restrict__ D,
    int M, int N, int Kd, float alpha,
    ll bA, ll rA, ll cA, ll bB, ll kB, ll nB, ll bD, ll rD, ll cD)
{
    __shared__ _Float16 As [64][48];   // 96B row stride, 16B aligned
    __shared__ _Float16 BsT[64][48];   // transposed B tile: [n][k]

    const int tid  = threadIdx.x;
    const int lane = tid & 31;
    const int wave = tid >> 5;        // 0..3 -> 16-row strip
    const int half = lane >> 4;       // 0/1
    const int l16  = lane & 15;

    const int m0 = blockIdx.y * 64;
    const int n0 = blockIdx.x * 64;
    const int bz = blockIdx.z;
    const ll  baseA = (ll)bz * bA;
    const ll  baseB = (ll)bz * bB;

    const v8f vzero = {0.f,0.f,0.f,0.f,0.f,0.f,0.f,0.f};
    v8f acc[4];
    #pragma unroll
    for (int j = 0; j < 4; ++j) acc[j] = vzero;

    for (int k0 = 0; k0 < Kd; k0 += 32) {
        // prefetch next A k-tile (speculative, one line per thread)
        if (k0 + 32 < Kd) {
            int pm = m0 + (tid & 63);
            if (pm < M)
                __builtin_prefetch((const void*)(A + baseA + (ll)pm*rA + (ll)(k0+32)*cA), 0, 1);
        }
        // ---- stage A tile 64x32: 4 floats/thread -> one ds_store_b64 ----
        for (int i = tid; i < 512; i += 128) {
            int r = i >> 3, c4 = (i & 7) << 2;
            int gm = m0 + r, gk = k0 + c4;
            float v0 = 0.f, v1 = 0.f, v2 = 0.f, v3 = 0.f;
            if (gm < M && gk < Kd) {
                const float* p = A + baseA + (ll)gm*rA + (ll)gk*cA;
                if (cA == 1) { v0 = p[0]; v1 = p[1]; v2 = p[2]; v3 = p[3]; }
                else         { v0 = p[0]; v1 = p[cA]; v2 = p[2*cA]; v3 = p[3*cA]; }
            }
            v4h pk; pk[0] = (_Float16)v0; pk[1] = (_Float16)v1;
                    pk[2] = (_Float16)v2; pk[3] = (_Float16)v3;
            *(v4h*)&As[r][c4] = pk;
        }
        // ---- stage B tile 32x64 transposed into BsT[n][k] ----
        for (int i = tid; i < 512; i += 128) {
            int n = i >> 3, k4 = (i & 7) << 2;
            int gn = n0 + n, gk = k0 + k4;
            float v0 = 0.f, v1 = 0.f, v2 = 0.f, v3 = 0.f;
            if (gn < N && gk < Kd) {
                const float* p = Bm + baseB + (ll)gk*kB + (ll)gn*nB;
                if (kB == 1) { v0 = p[0]; v1 = p[1]; v2 = p[2]; v3 = p[3]; }
                else         { v0 = p[0]; v1 = p[kB]; v2 = p[2*kB]; v3 = p[3*kB]; }
            }
            v4h pk; pk[0] = (_Float16)v0; pk[1] = (_Float16)v1;
                    pk[2] = (_Float16)v2; pk[3] = (_Float16)v3;
            *(v4h*)&BsT[n][k4] = pk;
        }
        __syncthreads();

        // ---- A fragment: two contiguous 16B LDS loads ----
        const int arow = wave*16 + l16;
        v8h alo = *(const v8h*)&As[arow][8*half];        // K = 8h..8h+7
        v8h ahi = *(const v8h*)&As[arow][16 + 8*half];   // K = 16+8h..
        v16h afrag = __builtin_shufflevector(alo, ahi,
                        0,1,2,3,4,5,6,7,8,9,10,11,12,13,14,15);

        #pragma unroll
        for (int j = 0; j < 4; ++j) {
            const int bcol = j*16 + l16;
            v8h blo = *(const v8h*)&BsT[bcol][8*half];       // K = 8h..8h+7
            v8h bhi = *(const v8h*)&BsT[bcol][16 + 8*half];  // K = 16+8h..
            v16h bfrag = __builtin_shufflevector(blo, bhi,
                            0,1,2,3,4,5,6,7,8,9,10,11,12,13,14,15);
            acc[j] = __builtin_amdgcn_wmma_f32_16x16x32_f16(
                         false, afrag, false, bfrag, (short)0, acc[j], false, false);
        }
        __syncthreads();
    }

    // epilogue (ISA C/D layout: VGPR v -> M = v + 8*half; N = lane&15)
    const ll baseD = (ll)bz * bD;
    #pragma unroll
    for (int j = 0; j < 4; ++j) {
        int gn = n0 + j*16 + l16;
        if (gn >= N) continue;
        float bv = bias ? bias[gn] : 0.f;
        #pragma unroll
        for (int v = 0; v < 8; ++v) {
            int gm = m0 + wave*16 + v + half*8;
            if (gm >= M) continue;
            float val = acc[j][v] * alpha + bv;
            if (ACT == ACT_RELU)      val = fmaxf(val, 0.f);
            else if (ACT == ACT_GELU) val = 0.5f * val * (1.f + erff(val * 0.70710678118654752f));
            ll off = baseD + (ll)gm*rD + (ll)gn*cD;
            if (Res)   val += Res[off];
            if (scale) { int si = bz*M + gm; val = val*scale[si] + shift[si]; }
            D[off] = val;
        }
    }
}

// ---------------- RevIN: per-(b,c) mean/std over L ----------------
__global__ __launch_bounds__(256) void revin_stats(const float* __restrict__ x,
                                                   float* mean, float* stdv)
{
    int r = blockIdx.x, b = r / C_, c = r % C_;
    const float* p = x + (ll)b*L_*C_ + c;
    float s = 0.f, ss = 0.f;
    for (int l = threadIdx.x; l < L_; l += 256) { float v = p[(ll)l*C_]; s += v; ss += v*v; }
    __shared__ float sh[256], sh2[256];
    sh[threadIdx.x] = s; sh2[threadIdx.x] = ss; __syncthreads();
    for (int o = 128; o > 0; o >>= 1) {
        if (threadIdx.x < o) { sh[threadIdx.x] += sh[threadIdx.x+o]; sh2[threadIdx.x] += sh2[threadIdx.x+o]; }
        __syncthreads();
    }
    if (threadIdx.x == 0) {
        float m = sh[0] / L_, v = sh2[0] / L_ - m*m;
        mean[r] = m; stdv[r] = sqrtf(fmaxf(v, 0.f) + EPSF);
    }
}

__global__ void revin_apply(const float* __restrict__ x, const float* __restrict__ mean,
                            const float* __restrict__ stdv, float* __restrict__ y, ll total)
{
    ll i = (ll)blockIdx.x*blockDim.x + threadIdx.x;
    if (i >= total) return;
    int c = (int)(i % C_);
    int b = (int)(i / ((ll)L_*C_));
    int r = b*C_ + c;
    y[i] = (x[i] - mean[r]) / stdv[r];
}

// ---------------- BatchNorm stats/apply ----------------
__global__ __launch_bounds__(256) void bn_t_stats(const float* __restrict__ x,
                                                  float* mean, float* invs)
{   // channels = L dim: stats over (B,C) per l
    int l = blockIdx.x;
    float s = 0.f, ss = 0.f;
    for (int i = threadIdx.x; i < B_*C_; i += 256) {
        int b = i / C_, c = i % C_;
        float v = x[(ll)b*L_*C_ + (ll)l*C_ + c];
        s += v; ss += v*v;
    }
    __shared__ float sh[256], sh2[256];
    sh[threadIdx.x] = s; sh2[threadIdx.x] = ss; __syncthreads();
    for (int o = 128; o > 0; o >>= 1) {
        if (threadIdx.x < o) { sh[threadIdx.x] += sh[threadIdx.x+o]; sh2[threadIdx.x] += sh2[threadIdx.x+o]; }
        __syncthreads();
    }
    if (threadIdx.x == 0) {
        float m = sh[0] / (B_*C_), v = sh2[0] / (B_*C_) - m*m;
        mean[l] = m; invs[l] = rsqrtf(fmaxf(v, 0.f) + EPSF);
    }
}

__global__ void bn_t_apply(const float* __restrict__ x, const float* __restrict__ mean,
                           const float* __restrict__ invs, const float* __restrict__ g,
                           const float* __restrict__ bt, float* __restrict__ y, ll total)
{
    ll i = (ll)blockIdx.x*blockDim.x + threadIdx.x;
    if (i >= total) return;
    int l = (int)((i / C_) % L_);
    y[i] = (x[i] - mean[l]) * invs[l] * g[l] + bt[l];
}

__global__ __launch_bounds__(256) void bn_f_stats(const float* __restrict__ x,
                                                  float* mean, float* invs)
{   // channels = C dim: stats over (B,L) per c
    int c = blockIdx.x;
    float s = 0.f, ss = 0.f;
    for (int i = threadIdx.x; i < B_*L_; i += 256) {
        float v = x[(ll)i*C_ + c];
        s += v; ss += v*v;
    }
    __shared__ float sh[256], sh2[256];
    sh[threadIdx.x] = s; sh2[threadIdx.x] = ss; __syncthreads();
    for (int o = 128; o > 0; o >>= 1) {
        if (threadIdx.x < o) { sh[threadIdx.x] += sh[threadIdx.x+o]; sh2[threadIdx.x] += sh2[threadIdx.x+o]; }
        __syncthreads();
    }
    if (threadIdx.x == 0) {
        float m = sh[0] / (B_*L_), v = sh2[0] / (B_*L_) - m*m;
        mean[c] = m; invs[c] = rsqrtf(fmaxf(v, 0.f) + EPSF);
    }
}

__global__ void bn_f_apply(const float* __restrict__ x, const float* __restrict__ mean,
                           const float* __restrict__ invs, const float* __restrict__ g,
                           const float* __restrict__ bt, float* __restrict__ y, ll total)
{
    ll i = (ll)blockIdx.x*blockDim.x + threadIdx.x;
    if (i >= total) return;
    int c = (int)(i % C_);
    y[i] = (x[i] - mean[c]) * invs[c] * g[c] + bt[c];
}

// ---------------- routing helpers ----------------
__global__ void rownorm_h(const float* __restrict__ h, float* __restrict__ den, ll rows)
{
    ll r = (ll)blockIdx.x*blockDim.x + threadIdx.x;
    if (r >= rows) return;
    const float* p = h + r*H_;
    float s = 0.f;
    for (int i = 0; i < H_; ++i) { float v = p[i]; s += v*v; }
    den[r] = sqrtf(s) + 1e-8f;
}

__global__ void cluster_norm(const float* __restrict__ ce, float* __restrict__ cnn)
{
    int k = blockIdx.x*blockDim.x + threadIdx.x;
    if (k >= K_) return;
    float s = 0.f;
    for (int i = 0; i < H_; ++i) { float v = ce[k*H_+i]; s += v*v; }
    float d = 1.f / (sqrtf(s) + 1e-8f);
    for (int i = 0; i < H_; ++i) cnn[k*H_+i] = ce[k*H_+i] * d;
}

// softmax over K=64 clusters; writes p (in place over logits) and mask
__global__ void route_softmax(float* __restrict__ S, const float* __restrict__ den,
                              const float* __restrict__ u, float* __restrict__ Mm, ll rows)
{
    ll r = (ll)blockIdx.x*blockDim.x + threadIdx.x;
    if (r >= rows) return;
    float d = 1.f / den[r];
    float* s = S + r*K_;
    float mx = -3.4e38f;
    for (int k = 0; k < K_; ++k) mx = fmaxf(mx, s[k]*d);
    float sum = 0.f;
    for (int k = 0; k < K_; ++k) sum += expf(s[k]*d - mx);
    float inv = 1.f / sum;
    const float* ur = u + r*K_;
    float* mr = Mm + r*K_;
    for (int k = 0; k < K_; ++k) {
        float p = expf(s[k]*d - mx) * inv;
        s[k] = p;
        mr[k] = (ur[k] < p) ? 1.f : 0.f;
    }
}

__global__ __launch_bounds__(256) void meanp_reduce(const float* __restrict__ P, float* __restrict__ mp)
{
    int r = blockIdx.x, b = r / K_, k = r % K_;
    float s = 0.f;
    for (int l = threadIdx.x; l < L_; l += 256)
        s += P[(ll)b*L_*K_ + (ll)l*K_ + k];
    __shared__ float sh[256];
    sh[threadIdx.x] = s; __syncthreads();
    for (int o = 128; o > 0; o >>= 1) {
        if (threadIdx.x < o) sh[threadIdx.x] += sh[threadIdx.x+o];
        __syncthreads();
    }
    if (threadIdx.x == 0) mp[r] = sh[0] / L_;
}

// aw = softmax_l( exp(scores) * mask ), in place over scores rows of length L
__global__ __launch_bounds__(256) void attn_softmax(float* __restrict__ Sc, const float* __restrict__ Mm)
{
    int r = blockIdx.x, b = r / K_, k = r % K_;
    float* row = Sc + (ll)r * L_;
    const float* mrow = Mm + (ll)b*L_*K_ + k;   // stride K_
    __shared__ float sh[256];
    float mx = -3.4e38f;
    for (int l = threadIdx.x; l < L_; l += 256)
        mx = fmaxf(mx, expf(row[l]) * mrow[(ll)l*K_]);
    sh[threadIdx.x] = mx; __syncthreads();
    for (int o = 128; o > 0; o >>= 1) {
        if (threadIdx.x < o) sh[threadIdx.x] = fmaxf(sh[threadIdx.x], sh[threadIdx.x+o]);
        __syncthreads();
    }
    mx = sh[0]; __syncthreads();
    float sum = 0.f;
    for (int l = threadIdx.x; l < L_; l += 256)
        sum += expf(expf(row[l]) * mrow[(ll)l*K_] - mx);
    sh[threadIdx.x] = sum; __syncthreads();
    for (int o = 128; o > 0; o >>= 1) {
        if (threadIdx.x < o) sh[threadIdx.x] += sh[threadIdx.x+o];
        __syncthreads();
    }
    float inv = 1.f / sh[0]; __syncthreads();
    for (int l = threadIdx.x; l < L_; l += 256) {
        float e = expf(row[l]) * mrow[(ll)l*K_];
        row[l] = expf(e - mx) * inv;
    }
}

__global__ void cluster_update(const float* __restrict__ ao, const float* __restrict__ mp,
                               float* __restrict__ outp)
{
    int t = blockIdx.x*blockDim.x + threadIdx.x;
    if (t >= K_*H_) return;
    int k = t / H_, hh = t % H_;
    float s = 0.f;
    for (int b = 0; b < B_; ++b)
        s += ao[(ll)b*K_*H_ + (ll)k*H_ + hh] * mp[b*K_ + k];
    outp[t] = s / B_;
}

// ---------------- host-side GEMM dispatcher ----------------
static void gemm(hipStream_t st, int act,
                 const float* A, const float* Bm, const float* bias,
                 const float* Res, const float* scale, const float* shift,
                 float* D, int M, int N, int Kd, float alpha, int batch,
                 ll bA, ll rA, ll cA, ll bB, ll kB, ll nB, ll bD, ll rD, ll cD)
{
    dim3 grid((N + 63) / 64, (M + 63) / 64, batch), block(128);
    switch (act) {
    case ACT_RELU:
        gemm_wmma<ACT_RELU><<<grid, block, 0, st>>>(A,Bm,bias,Res,scale,shift,D,M,N,Kd,alpha,bA,rA,cA,bB,kB,nB,bD,rD,cD);
        break;
    case ACT_GELU:
        gemm_wmma<ACT_GELU><<<grid, block, 0, st>>>(A,Bm,bias,Res,scale,shift,D,M,N,Kd,alpha,bA,rA,cA,bB,kB,nB,bD,rD,cD);
        break;
    default:
        gemm_wmma<ACT_NONE><<<grid, block, 0, st>>>(A,Bm,bias,Res,scale,shift,D,M,N,Kd,alpha,bA,rA,cA,bB,kB,nB,bD,rD,cD);
        break;
    }
}

extern "C" void kernel_launch(void* const* d_in, const int* in_sizes, int n_in,
                              void* d_out, int out_size, void* d_ws, size_t ws_size,
                              hipStream_t stream)
{
    (void)in_sizes; (void)n_in; (void)out_size; (void)ws_size;

    const float* x     = (const float*)d_in[0];
    const float* uber  = (const float*)d_in[1];
    const float* ce    = (const float*)d_in[2];
    const float* wq_w  = (const float*)d_in[3],  *wq_b  = (const float*)d_in[4];
    const float* wk_w  = (const float*)d_in[5],  *wk_b  = (const float*)d_in[6];
    const float* wv_w  = (const float*)d_in[7],  *wv_b  = (const float*)d_in[8];
    const float* cm_w1 = (const float*)d_in[9],  *cm_b1 = (const float*)d_in[10];
    const float* cm_w2 = (const float*)d_in[11], *cm_b2 = (const float*)d_in[12];
    const float* tm_w1 = (const float*)d_in[13], *tm_b1 = (const float*)d_in[14];
    const float* tm_w2 = (const float*)d_in[15], *tm_b2 = (const float*)d_in[16];
    const float* bntg  = (const float*)d_in[17], *bntb  = (const float*)d_in[18];
    const float* ltw   = (const float*)d_in[19], *ltb   = (const float*)d_in[20];
    const float* bnfg  = (const float*)d_in[21], *bnfb  = (const float*)d_in[22];
    const float* lf1w  = (const float*)d_in[23], *lf1b  = (const float*)d_in[24];
    const float* lf2w  = (const float*)d_in[25], *lf2b  = (const float*)d_in[26];
    const float* outw  = (const float*)d_in[27], *outb  = (const float*)d_in[28];

    // ---- workspace carving ----
    float* wsp = (float*)d_ws;
    size_t off = 0;
    auto alloc = [&](size_t n) { float* p = wsp + off; off += n; return p; };
    const size_t BLC = (size_t)B_ * L_ * C_;

    float* mean1 = alloc(B_*C_);        float* std1  = alloc(B_*C_);
    float* mean2 = alloc(B_*C_);        float* std2  = alloc(B_*C_);
    float* meanL = alloc(L_);           float* invL  = alloc(L_);
    float* meanC = alloc(C_);           float* invC  = alloc(C_);
    float* hden  = alloc((size_t)B_*L_);
    float* cnn   = alloc(K_*H_);
    float* qm    = alloc(K_*H_);
    float* mp    = alloc(B_*K_);
    float* ao    = alloc((size_t)B_*K_*H_);
    float* S     = alloc((size_t)B_*L_*K_);   // routing logits -> p
    float* Mm    = alloc((size_t)B_*L_*K_);   // bernoulli mask
    float* sc    = alloc((size_t)B_*K_*L_);   // attn scores -> aw
    float* xn    = alloc(BLC);                // xn, later reused as Km
    float* t1    = alloc(BLC);                // scratch, later V / bn output
    float* h     = alloc(BLC);
    float* t2    = alloc(BLC);                // scratch
    float* zb    = alloc(BLC);                // h2 -> z

    const int th = 256;
    const ll rows = (ll)B_ * L_;

    // 1) RevIN norm of x
    revin_stats<<<B_*C_, th, 0, stream>>>(x, mean1, std1);
    revin_apply<<<(int)((BLC + 255) / 256), th, 0, stream>>>(x, mean1, std1, xn, (ll)BLC);

    // 2) channel MLP: h = relu(xn@cm_w1+b1)@cm_w2+b2
    gemm(stream, ACT_RELU, xn, cm_w1, cm_b1, nullptr, nullptr, nullptr, t1,
         B_*L_, H_, C_, 1.f, 1, 0, C_, 1, 0, H_, 1, 0, H_, 1);
    gemm(stream, ACT_NONE, t1, cm_w2, cm_b2, nullptr, nullptr, nullptr, h,
         B_*L_, H_, H_, 1.f, 1, 0, H_, 1, 0, H_, 1, 0, H_, 1);

    // 3) cosine routing: S = h @ cnn^T, softmax/K, Bernoulli mask, mean-p
    rownorm_h<<<(int)((rows + 255) / 256), th, 0, stream>>>(h, hden, rows);
    cluster_norm<<<1, 64, 0, stream>>>(ce, cnn);
    gemm(stream, ACT_NONE, h, cnn, nullptr, nullptr, nullptr, nullptr, S,
         B_*L_, K_, H_, 1.f, 1, 0, H_, 1, 0, 1, H_, 0, K_, 1);          // B^T via strides
    route_softmax<<<(int)((rows + 255) / 256), th, 0, stream>>>(S, hden, uber, Mm, rows);
    meanp_reduce<<<B_*K_, th, 0, stream>>>(S, mp);

    // 4) Q/K/V projections (Km -> xn, V -> t1; xn/t1 free now)
    gemm(stream, ACT_NONE, h, wk_w, wk_b, nullptr, nullptr, nullptr, xn,
         B_*L_, H_, H_, 1.f, 1, 0, H_, 1, 0, H_, 1, 0, H_, 1);
    gemm(stream, ACT_NONE, h, wv_w, wv_b, nullptr, nullptr, nullptr, t1,
         B_*L_, H_, H_, 1.f, 1, 0, H_, 1, 0, H_, 1, 0, H_, 1);
    gemm(stream, ACT_NONE, ce, wq_w, wq_b, nullptr, nullptr, nullptr, qm,
         K_, H_, H_, 1.f, 1, 0, H_, 1, 0, H_, 1, 0, H_, 1);

    // 5) scores = (Q @ Km^T)/sqrt(H), batched over B
    gemm(stream, ACT_NONE, qm, xn, nullptr, nullptr, nullptr, nullptr, sc,
         K_, L_, H_, 0.08838834764831845f, B_,
         0, H_, 1, (ll)L_*H_, 1, H_, (ll)K_*L_, L_, 1);
    attn_softmax<<<B_*K_, th, 0, stream>>>(sc, Mm);

    // 6) ao = aw @ V (batched), then cluster update into d_out tail
    gemm(stream, ACT_NONE, sc, t1, nullptr, nullptr, nullptr, nullptr, ao,
         K_, H_, L_, 1.f, B_,
         (ll)K_*L_, L_, 1, (ll)L_*H_, H_, 1, (ll)K_*H_, H_, 1);
    cluster_update<<<(K_*H_ + 255) / 256, th, 0, stream>>>(ao, mp, (float*)d_out + (size_t)B_*P_*C_);

    // 7) temporal MLP: h2 = relu(h@tm_w1+b1)@tm_w2+b2  -> zb
    gemm(stream, ACT_RELU, h, tm_w1, tm_b1, nullptr, nullptr, nullptr, t2,
         B_*L_, H_, H_, 1.f, 1, 0, H_, 1, 0, H_, 1, 0, H_, 1);
    gemm(stream, ACT_NONE, t2, tm_w2, tm_b2, nullptr, nullptr, nullptr, zb,
         B_*L_, H_, H_, 1.f, 1, 0, H_, 1, 0, H_, 1, 0, H_, 1);

    // 8) second RevIN (stats kept for denorm), normalize in place -> z
    revin_stats<<<B_*C_, th, 0, stream>>>(zb, mean2, std2);
    revin_apply<<<(int)((BLC + 255) / 256), th, 0, stream>>>(zb, mean2, std2, zb, (ll)BLC);

    // 9) 4 mixer blocks
    for (int it = 0; it < NBLK; ++it) {
        // timestep mixer: BN over L-channels, GEMM over seq dim (implicit transpose)
        bn_t_stats<<<L_, th, 0, stream>>>(zb, meanL, invL);
        bn_t_apply<<<(int)((BLC + 255) / 256), th, 0, stream>>>(zb, meanL, invL, bntg, bntb, t1, (ll)BLC);
        // z = z + gelu( z^T @ lin_t_w + b )^T     (A/D strides give the swapaxes view)
        gemm(stream, ACT_GELU, t1, ltw, ltb, zb, nullptr, nullptr, zb,
             C_, L_, L_, 1.f, B_,
             (ll)L_*C_, 1, C_, 0, L_, 1, (ll)L_*C_, 1, C_);
        // feature mixer
        bn_f_stats<<<C_, th, 0, stream>>>(zb, meanC, invC);
        bn_f_apply<<<(int)((BLC + 255) / 256), th, 0, stream>>>(zb, meanC, invC, bnfg, bnfb, t1, (ll)BLC);
        gemm(stream, ACT_GELU, t1, lf1w, lf1b, nullptr, nullptr, nullptr, t2,
             B_*L_, H_, C_, 1.f, 1, 0, C_, 1, 0, H_, 1, 0, H_, 1);
        gemm(stream, ACT_NONE, t2, lf2w, lf2b, zb, nullptr, nullptr, zb,
             B_*L_, C_, H_, 1.f, 1, 0, H_, 1, 0, C_, 1, 0, C_, 1);
    }

    // 10) head: y[b,p,c] = (z^T @ out_w + out_b)[b,c,p] * std2[b,c] + mean2[b,c]
    gemm(stream, ACT_NONE, zb, outw, outb, nullptr, std2, mean2, (float*)d_out,
         C_, P_, L_, 1.f, B_,
         (ll)L_*C_, 1, C_, 0, P_, 1, (ll)P_*C_, 1, C_);
}
